// Critic_mean_83124797046898
// MI455X (gfx1250) — matrix operand order repaired and checked by reading, not compiled, not run on hardware
//
#include <hip/hip_runtime.h>
#include <hip/hip_bf16.h>

// ---------------------------------------------------------------------------
// Gasse bipartite GNN critic on MI455X (gfx1250).
//   EMB=64, N_C=N_V=64000, E=1,024,000.
// All 64-wide GEMMs use v_wmma_f32_16x16x32_bf16 (bf16 in, f32 accum).
// NOTE: LN over a single edge feature is identically zero, so the e@we term
// is dropped (mathematically exact, not an approximation).
// ---------------------------------------------------------------------------

typedef __attribute__((ext_vector_type(16))) __bf16 v16bf;
typedef __attribute__((ext_vector_type(2)))  __bf16 v2bf;
typedef __attribute__((ext_vector_type(8)))  float  v8f;
typedef unsigned short u16;
typedef unsigned int   u32;

union Frag16 { v16bf v; u16 u[16]; u32 d[8]; uint4 q[2]; };

__device__ __forceinline__ u16 f2bf(float x) {
  u32 u = __float_as_uint(x);
  u32 r = u + 0x7fffu + ((u >> 16) & 1u);   // round-to-nearest-even
  return (u16)(r >> 16);
}
__device__ __forceinline__ float bf2f(u16 h) {
  return __uint_as_float(((u32)h) << 16);
}
// Packed pair convert -> one DWORD (lo -> bits[15:0]).
// Uses v_cvt_pk_bf16_f32 when the builtin exists; manual RNE otherwise.
__device__ __forceinline__ u32 pk_bf16(float lo, float hi) {
#if __has_builtin(__builtin_amdgcn_cvt_pk_bf16_f32)
  v2bf p = __builtin_amdgcn_cvt_pk_bf16_f32(lo, hi);
  u32 d;
  __builtin_memcpy(&d, &p, 4);
  return d;
#else
  return (u32)f2bf(lo) | ((u32)f2bf(hi) << 16);
#endif
}

// ---------------------------------------------------------------------------
// Weight prep: W[K,64] f32 row-major -> Wt[64,K] bf16 (column of W contiguous)
// ---------------------------------------------------------------------------
__global__ void prep_wt(const float* __restrict__ W, u16* __restrict__ Wt, int K) {
  int t = blockIdx.x * blockDim.x + threadIdx.x;
  if (t >= 64 * K) return;
  int n = t / K, k = t % K;
  Wt[n * K + k] = f2bf(W[k * 64 + n]);
}

// ---------------------------------------------------------------------------
// LayerNorm + first MLP layer (K=5 or 19, too skinny for WMMA): one wave/row.
// H[row][j] = relu( sum_k ln(x)[k] * W1[k][j] + b1[j] ), bf16 out.
// ---------------------------------------------------------------------------
__global__ void ln_mlp1(const float* __restrict__ X, const float* __restrict__ W1,
                        const float* __restrict__ b1, u16* __restrict__ H,
                        int N, int K) {
  int lane = threadIdx.x & 31;
  int row  = blockIdx.x * (blockDim.x >> 5) + (threadIdx.x >> 5);
  if (row >= N) return;
  const float* x = X + (size_t)row * K;
  float mean = 0.f;
  for (int k = 0; k < K; ++k) mean += x[k];        // wave-uniform addrs: broadcast
  mean /= (float)K;
  float var = 0.f;
  for (int k = 0; k < K; ++k) { float d = x[k] - mean; var += d * d; }
  var /= (float)K;
  float rs = rsqrtf(var + 1e-5f);
  #pragma unroll
  for (int h = 0; h < 2; ++h) {
    int j = lane + h * 32;
    float acc = b1[j];
    for (int k = 0; k < K; ++k) acc += (x[k] - mean) * rs * W1[k * 64 + j];
    H[(size_t)row * 64 + j] = f2bf(fmaxf(acc, 0.f));
  }
}

// ---------------------------------------------------------------------------
// Generic GEMM: Y[N,64] = act(X[N,K]bf16 @ W[K,64] + bias), K in {64,128}.
// Wt is the bf16 [64,K] transposed weight; staged in LDS per block.
// One wave computes a 16x64 output tile = 4 WMMA column tiles, K/32 k-steps
// (fully unrolled via template K). Fragment layouts per CDNA5 ISA 7.12.2.
// ---------------------------------------------------------------------------
template <int K, bool RELU, bool OUT_BF16>
__global__ void gemm_n64(const u16* __restrict__ X, const u16* __restrict__ Wt,
                         const float* __restrict__ bias,
                         u16* __restrict__ Yb, float* __restrict__ Yf,
                         int N) {
  __shared__ __align__(16) u16 sW[64 * K];
  int tid = threadIdx.x;
  for (int i = tid; i < (64 * K) >> 1; i += blockDim.x)
    ((u32*)sW)[i] = ((const u32*)Wt)[i];
  __syncthreads();

  int lane = tid & 31, wv = tid >> 5;
  int row0 = (blockIdx.x * (blockDim.x >> 5) + wv) * 16;
  if (row0 >= N) return;

  int m = lane & 15, half = lane >> 4;
  const u16* xrow = X + (size_t)(row0 + m) * K;
  v8f acc[4];
  #pragma unroll
  for (int nt = 0; nt < 4; ++nt) acc[nt] = (v8f){0,0,0,0,0,0,0,0};

  #pragma unroll
  for (int kt = 0; kt < K; kt += 32) {
    Frag16 a;
    int o0 = kt + half * 8;                        // lane-lo: K 0..7,16..23; hi: +8
    a.q[0] = *(const uint4*)(xrow + o0);
    a.q[1] = *(const uint4*)(xrow + o0 + 16);
    #pragma unroll
    for (int nt = 0; nt < 4; ++nt) {
      Frag16 b;
      const u16* wr = sW + (size_t)(m + nt * 16) * K + kt + half * 8;
      b.q[0] = *(const uint4*)(wr);
      b.q[1] = *(const uint4*)(wr + 16);
      acc[nt] = __builtin_amdgcn_wmma_f32_16x16x32_bf16(
          false, a.v, false, b.v, (short)0, acc[nt], false, false);
    }
  }
  // C layout: VGPR r -> row (half*8 + r), col = (lane&15) + nt*16
  #pragma unroll
  for (int nt = 0; nt < 4; ++nt) {
    int n = m + nt * 16;
    float bv = bias ? bias[n] : 0.f;
    #pragma unroll
    for (int r = 0; r < 8; ++r) {
      float v = acc[nt][r] + bv;
      if (RELU) v = fmaxf(v, 0.f);
      size_t idx = (size_t)(row0 + half * 8 + r) * 64 + n;
      if (OUT_BF16) Yb[idx] = f2bf(v);
      else          Yf[idx] = v;
    }
  }
}

// ---------------------------------------------------------------------------
// Edge message kernel: 16 edges per wave.
//   joint = srcp[si[e]] + dstp[di[e]]  (f32 gathers, L2-resident tables)
//   msg   = relu(joint) @ wf + bf      (2 k-steps x 4 n-tiles of WMMA)
//   agg[di[e]] += msg                  (global_atomic_add_f32 scatter)
// A-fragment built with packed bf16 converts (one DWORD per 2 elems).
// ---------------------------------------------------------------------------
__global__ void edge_msg(const float* __restrict__ srcp, const float* __restrict__ dstp,
                         const int* __restrict__ sidx, const int* __restrict__ didx,
                         const u16* __restrict__ Wt,   // wf^T bf16 [64,64]
                         const float* __restrict__ bf,
                         float* __restrict__ agg, int E) {
  __shared__ __align__(16) u16 sW[64 * 64];
  __shared__ float sB[64];
  int tid = threadIdx.x;
  for (int i = tid; i < (64 * 64) >> 1; i += blockDim.x)
    ((u32*)sW)[i] = ((const u32*)Wt)[i];
  if (tid < 64) sB[tid] = bf[tid];
  __syncthreads();

  int lane = tid & 31, wv = tid >> 5;
  int e0 = (blockIdx.x * (blockDim.x >> 5) + wv) * 16;
  if (e0 >= E) return;

  int m = lane & 15, half = lane >> 4;
  int e  = min(e0 + m, E - 1);
  __builtin_prefetch(sidx + e + 2048, 0, 0);       // global_prefetch_b8 on idx stream
  int si = sidx[e], di = didx[e];
  const float* sr = srcp + (size_t)si * 64;
  const float* dr = dstp + (size_t)di * 64;

  v8f acc[4];
  #pragma unroll
  for (int nt = 0; nt < 4; ++nt) acc[nt] = (v8f){0,0,0,0,0,0,0,0};

  #pragma unroll
  for (int kt = 0; kt < 64; kt += 32) {
    Frag16 a;
    int o0 = kt + half * 8;
    #pragma unroll
    for (int blk = 0; blk < 2; ++blk) {            // K o0..o0+7 then o0+16..o0+23
      int ko = o0 + blk * 16;
      float4 s0 = *(const float4*)(sr + ko);
      float4 s1 = *(const float4*)(sr + ko + 4);
      float4 d0 = *(const float4*)(dr + ko);
      float4 d1 = *(const float4*)(dr + ko + 4);
      a.d[blk*4+0] = pk_bf16(fmaxf(s0.x + d0.x, 0.f), fmaxf(s0.y + d0.y, 0.f));
      a.d[blk*4+1] = pk_bf16(fmaxf(s0.z + d0.z, 0.f), fmaxf(s0.w + d0.w, 0.f));
      a.d[blk*4+2] = pk_bf16(fmaxf(s1.x + d1.x, 0.f), fmaxf(s1.y + d1.y, 0.f));
      a.d[blk*4+3] = pk_bf16(fmaxf(s1.z + d1.z, 0.f), fmaxf(s1.w + d1.w, 0.f));
    }
    #pragma unroll
    for (int nt = 0; nt < 4; ++nt) {
      Frag16 b;
      const u16* wr = sW + (size_t)(m + nt * 16) * 64 + kt + half * 8;
      b.q[0] = *(const uint4*)(wr);
      b.q[1] = *(const uint4*)(wr + 16);
      acc[nt] = __builtin_amdgcn_wmma_f32_16x16x32_bf16(
          false, a.v, false, b.v, (short)0, acc[nt], false, false);
    }
  }
  // scatter-add: row (half*8+r), col (lane&15)+nt*16
  #pragma unroll
  for (int r = 0; r < 8; ++r) {
    int ee = e0 + half * 8 + r;
    if (ee < E) {
      int d = didx[ee];
      float* dst = agg + (size_t)d * 64 + m;
      #pragma unroll
      for (int nt = 0; nt < 4; ++nt)
        atomicAdd(dst + nt * 16, acc[nt][r] + sB[m + nt * 16]);
    }
  }
}

// ---------------------------------------------------------------------------
// H[i] = [ bf16(agg[i]) || dst_emb[i] ]  -> [N,128] bf16.
// One thread handles an aligned pair: packed convert + DWORD stores.
// ---------------------------------------------------------------------------
__global__ void concat_h(const float* __restrict__ agg, const u32* __restrict__ demb,
                         u32* __restrict__ H, int N) {
  int t = blockIdx.x * blockDim.x + threadIdx.x;
  if (t >= N * 32) return;
  int i = t >> 5, jp = t & 31;                      // jp: pair index 0..31
  float2 a = *(const float2*)(agg + (size_t)i * 64 + jp * 2);
  H[(size_t)i * 64 + jp]      = pk_bf16(a.x, a.y);  // row stride = 64 DWORDs
  H[(size_t)i * 64 + 32 + jp] = demb[(size_t)i * 32 + jp];
}

// ---------------------------------------------------------------------------
// Head: out[g] += (T1[row]·ow2 + ob2)/vars_per_g   (one wave per row)
// ---------------------------------------------------------------------------
__global__ void head_reduce(const u16* __restrict__ T1, const float* __restrict__ ow2,
                            const float* __restrict__ ob2, float* __restrict__ out,
                            int N, int vars_per_g) {
  int lane = threadIdx.x & 31;
  int row  = blockIdx.x * (blockDim.x >> 5) + (threadIdx.x >> 5);
  if (row >= N) return;
  const u16* t = T1 + (size_t)row * 64;
  float v = bf2f(t[lane]) * ow2[lane] + bf2f(t[lane + 32]) * ow2[lane + 32];
  for (int o = 16; o > 0; o >>= 1) v += __shfl_down(v, o, 32);
  if (lane == 0)
    atomicAdd(&out[row / vars_per_g], (v + ob2[0]) / (float)vars_per_g);
}

// ---------------------------------------------------------------------------
// Host-side orchestration
// ---------------------------------------------------------------------------
static inline int cdiv(int a, int b) { return (a + b - 1) / b; }

extern "C" void kernel_launch(void* const* d_in, const int* in_sizes, int n_in,
                              void* d_out, int out_size, void* d_ws, size_t ws_size,
                              hipStream_t stream) {
  // --- input mapping: top-level dict in insertion order; nested 'params'
  //     flattened as a JAX pytree (sorted keys, recursively). ---
  const float* cfeat = (const float*)d_in[0];            // [N_C,5]
  const float* vfeat = (const float*)d_in[2];            // [N_V,19]
  const float* c_b1  = (const float*)d_in[3];
  const float* c_b2  = (const float*)d_in[4];
  const float* c_w1  = (const float*)d_in[5];
  const float* c_w2  = (const float*)d_in[6];
  // conv_c2v: bf,o1b,o1w,o2b,o2w,we,wf,wl,wr  -> 7..15
  const float* c2v_bf  = (const float*)d_in[7];
  const float* c2v_o1b = (const float*)d_in[8];
  const float* c2v_o1w = (const float*)d_in[9];
  const float* c2v_o2b = (const float*)d_in[10];
  const float* c2v_o2w = (const float*)d_in[11];
  const float* c2v_wf  = (const float*)d_in[13];
  const float* c2v_wl  = (const float*)d_in[14];
  const float* c2v_wr  = (const float*)d_in[15];
  // conv_v2c -> 16..24
  const float* v2c_bf  = (const float*)d_in[16];
  const float* v2c_o1b = (const float*)d_in[17];
  const float* v2c_o1w = (const float*)d_in[18];
  const float* v2c_o2b = (const float*)d_in[19];
  const float* v2c_o2w = (const float*)d_in[20];
  const float* v2c_wf  = (const float*)d_in[22];
  const float* v2c_wl  = (const float*)d_in[23];
  const float* v2c_wr  = (const float*)d_in[24];
  const float* ob1 = (const float*)d_in[25];
  const float* ob2 = (const float*)d_in[26];
  const float* ow1 = (const float*)d_in[27];
  const float* ow2 = (const float*)d_in[28];
  const float* v_b1 = (const float*)d_in[29];
  const float* v_b2 = (const float*)d_in[30];
  const float* v_w1 = (const float*)d_in[31];
  const float* v_w2 = (const float*)d_in[32];
  const int* eidx = (const int*)d_in[33];

  const int N_C = in_sizes[0] / 5;
  const int N_V = in_sizes[2] / 19;
  const int E   = in_sizes[1];
  const int* ci = eidx;        // edge_index[0]
  const int* vi = eidx + E;    // edge_index[1]
  float* out = (float*)d_out;
  const int vars_per_g = N_V / (out_size > 0 ? out_size : 1);

  // --- workspace carve-up ---
  size_t off = 0;
  auto alloc = [&](size_t bytes) -> void* {
    void* p = (char*)d_ws + off;
    off = (off + bytes + 255) & ~(size_t)255;
    return p;
  };
  auto wt64  = [&]() { return (u16*)alloc(64 * 64 * sizeof(u16)); };
  u16* wt_cw2 = wt64();  u16* wt_vw2 = wt64();  u16* wt_ow1 = wt64();
  u16* wt_v2c_wl = wt64(); u16* wt_v2c_wr = wt64(); u16* wt_v2c_wf = wt64();
  u16* wt_v2c_o1w = (u16*)alloc(64 * 128 * sizeof(u16)); u16* wt_v2c_o2w = wt64();
  u16* wt_c2v_wl = wt64(); u16* wt_c2v_wr = wt64(); u16* wt_c2v_wf = wt64();
  u16* wt_c2v_o1w = (u16*)alloc(64 * 128 * sizeof(u16)); u16* wt_c2v_o2w = wt64();
  const int NMAX = (N_C > N_V) ? N_C : N_V;
  u16*   hid   = (u16*)alloc((size_t)NMAX * 64 * sizeof(u16));
  u16*   c_emb = (u16*)alloc((size_t)N_C * 64 * sizeof(u16));
  u16*   v_emb = (u16*)alloc((size_t)N_V * 64 * sizeof(u16));
  float* srcp  = (float*)alloc((size_t)NMAX * 64 * sizeof(float));
  float* dstp  = (float*)alloc((size_t)NMAX * 64 * sizeof(float));
  float* agg   = (float*)alloc((size_t)NMAX * 64 * sizeof(float));
  u16*   Hcat  = (u16*)alloc((size_t)NMAX * 128 * sizeof(u16));
  u16*   t1    = (u16*)alloc((size_t)NMAX * 64 * sizeof(u16));
  u16*   c_new = (u16*)alloc((size_t)N_C * 64 * sizeof(u16));
  u16*   v_new = (u16*)alloc((size_t)N_V * 64 * sizeof(u16));
  (void)ws_size; (void)n_in;

  const dim3 B256(256);
  auto prep = [&](const float* W, u16* Wt, int K) {
    prep_wt<<<cdiv(64 * K, 256), B256, 0, stream>>>(W, Wt, K);
  };
  prep(c_w2, wt_cw2, 64);   prep(v_w2, wt_vw2, 64);   prep(ow1, wt_ow1, 64);
  prep(v2c_wl, wt_v2c_wl, 64); prep(v2c_wr, wt_v2c_wr, 64); prep(v2c_wf, wt_v2c_wf, 64);
  prep(v2c_o1w, wt_v2c_o1w, 128); prep(v2c_o2w, wt_v2c_o2w, 64);
  prep(c2v_wl, wt_c2v_wl, 64); prep(c2v_wr, wt_c2v_wr, 64); prep(c2v_wf, wt_c2v_wf, 64);
  prep(c2v_o1w, wt_c2v_o1w, 128); prep(c2v_o2w, wt_c2v_o2w, 64);

  auto gemm_bf64 = [&](const u16* X, const u16* Wt, const float* bias, u16* Y, int N) {
    gemm_n64<64, true, true><<<cdiv(N / 16, 8), B256, 0, stream>>>(X, Wt, bias, Y, nullptr, N);
  };
  auto gemm_bf128 = [&](const u16* X, const u16* Wt, const float* bias, u16* Y, int N) {
    gemm_n64<128, true, true><<<cdiv(N / 16, 8), B256, 0, stream>>>(X, Wt, bias, Y, nullptr, N);
  };
  auto gemm_f32 = [&](const u16* X, const u16* Wt, float* Y, int N) {
    gemm_n64<64, false, false><<<cdiv(N / 16, 8), B256, 0, stream>>>(X, Wt, nullptr, nullptr, Y, N);
  };

  // --- node encoders ---
  ln_mlp1<<<cdiv(N_C, 8), B256, 0, stream>>>(cfeat, c_w1, c_b1, hid, N_C, 5);
  gemm_bf64(hid, wt_cw2, c_b2, c_emb, N_C);
  ln_mlp1<<<cdiv(N_V, 8), B256, 0, stream>>>(vfeat, v_w1, v_b1, hid, N_V, 19);
  gemm_bf64(hid, wt_vw2, v_b2, v_emb, N_V);

  // --- half-convolution helper ---
  auto halfconv = [&](const u16* semb, const u16* demb, int Ns, int Nd,
                      const int* si, const int* di,
                      u16* wl_t, u16* wr_t, u16* wf_t, const float* bf,
                      u16* o1w_t, const float* o1b, u16* o2w_t, const float* o2b,
                      u16* newd) {
    gemm_f32(semb, wl_t, srcp, Ns);
    gemm_f32(demb, wr_t, dstp, Nd);
    (void)hipMemsetAsync(agg, 0, (size_t)Nd * 64 * sizeof(float), stream);
    edge_msg<<<cdiv(E / 16, 8), B256, 0, stream>>>(srcp, dstp, si, di, wf_t, bf, agg, E);
    concat_h<<<cdiv(Nd * 32, 256), B256, 0, stream>>>(agg, (const u32*)demb, (u32*)Hcat, Nd);
    gemm_bf128(Hcat, o1w_t, o1b, t1, Nd);
    gemm_bf64(t1, o2w_t, o2b, newd, Nd);
  };

  // v -> c, then c -> v
  halfconv(v_emb, c_emb, N_V, N_C, vi, ci,
           wt_v2c_wl, wt_v2c_wr, wt_v2c_wf, v2c_bf,
           wt_v2c_o1w, v2c_o1b, wt_v2c_o2w, v2c_o2b, c_new);
  halfconv(c_new, v_emb, N_C, N_V, ci, vi,
           wt_c2v_wl, wt_c2v_wr, wt_c2v_wf, c2v_bf,
           wt_c2v_o1w, c2v_o1b, wt_c2v_o2w, c2v_o2b, v_new);

  // --- output head + per-graph mean ---
  gemm_bf64(v_new, wt_ow1, ob1, t1, N_V);
  (void)hipMemsetAsync(out, 0, (size_t)out_size * sizeof(float), stream);
  head_reduce<<<cdiv(N_V, 8), B256, 0, stream>>>(t1, ow2, ob2, out, N_V, vars_per_g);
}
